// BiasedMultiHeadAttention_89386859364537
// MI455X (gfx1250) — matrix operand order repaired and checked by reading, not compiled, not run on hardware
//
#include <hip/hip_runtime.h>
#include <hip/hip_bf16.h>

// ---- shapes (fixed by the reference) ----
#define BB 4
#define LL 1024
#define EE 1024
#define HH 16
#define DD 64
#define MM (BB * LL)       // 4096 tokens
#define LOG2E 1.44269504088896340736f

typedef __attribute__((ext_vector_type(16))) __bf16 v16bf;
typedef __attribute__((ext_vector_type(8)))  __bf16 v8bf;
typedef __attribute__((ext_vector_type(8)))  float  v8f;
typedef __attribute__((ext_vector_type(4)))  unsigned int u32x4;
typedef __attribute__((ext_vector_type(8)))  int i32x8;
typedef __attribute__((ext_vector_type(4)))  int i32x4;

#ifndef __has_builtin
#define __has_builtin(x) 0
#endif
#if __has_builtin(__builtin_amdgcn_tensor_load_to_lds)
#define HAVE_TDM 1
#else
#define HAVE_TDM 0
#endif

static __device__ inline __bf16 f2bf(float f) {
    unsigned u = __builtin_bit_cast(unsigned, f);
    unsigned r = u + 0x7FFFu + ((u >> 16) & 1u);
    unsigned short hs = (unsigned short)(r >> 16);
    return __builtin_bit_cast(__bf16, hs);
}

static __device__ inline v8f zero8() {
    v8f z;
#pragma unroll
    for (int i = 0; i < 8; ++i) z[i] = 0.0f;
    return z;
}

static __device__ inline v16bf mk16(v8bf lo, v8bf hi) {
    v16bf r;
#pragma unroll
    for (int i = 0; i < 8; ++i) { r[i] = lo[i]; r[i + 8] = hi[i]; }
    return r;
}

static __device__ inline v8f wmma_bf16(v16bf a, v16bf b, v8f c) {
    // D = A(16x32 bf16) * B(32x16 bf16) + C(16x16 f32)
    return __builtin_amdgcn_wmma_f32_16x16x32_bf16(
        false, a, false, b, (short)0, c, false, false);
}

#if HAVE_TDM
// ---- Tensor Data Mover descriptor helpers (D# per CDNA5 ISA) ----
static __device__ inline unsigned ldsaddr(const void* p) {
    // flat shared pointer: low 32 bits are the LDS byte address
    return (unsigned)(unsigned long long)(uintptr_t)p;
}

static __device__ inline u32x4 tdm_g0(unsigned lds_addr, unsigned long long gaddr) {
    u32x4 g;
    g[0] = 1u;                                                // count=1, user mode
    g[1] = lds_addr;                                          // LDS byte address
    g[2] = (unsigned)(gaddr & 0xFFFFFFFFull);                 // global_addr[31:0]
    g[3] = (unsigned)((gaddr >> 32) & 0x1FFFFFFull) | (2u << 30); // addr[56:32], type=2
    return g;
}

// 2-D tile, data_size code 2 (=4 bytes). dims/strides in dwords.
static __device__ inline i32x8 tdm_g1(unsigned tensor_d0, unsigned tensor_d1,
                                      unsigned tile_d0, unsigned tile_d1,
                                      unsigned long long d0_stride) {
    i32x8 g;
    g[0] = (int)(2u << 16);                                   // data_size = 4B
    g[1] = (int)((tensor_d0 & 0xFFFFu) << 16);                // tensor_dim0[15:0]
    g[2] = (int)((tensor_d0 >> 16) | ((tensor_d1 & 0xFFFFu) << 16));
    g[3] = (int)((tensor_d1 >> 16) | (tile_d0 << 16));        // tile_dim0
    g[4] = (int)(tile_d1 & 0xFFFFu);                          // tile_dim1 (tile_dim2=0)
    g[5] = (int)(d0_stride & 0xFFFFFFFFull);                  // tensor_dim0_stride lo
    g[6] = (int)((d0_stride >> 32) & 0xFFFFull);              // stride hi; dim1_stride=0
    g[7] = 0;
    return g;
}

static __device__ inline void tdm_load(u32x4 g0, i32x8 g1) {
    i32x4 z4 = {0, 0, 0, 0};
#if defined(__clang_major__) && (__clang_major__ >= 23)
    i32x8 z8 = {0, 0, 0, 0, 0, 0, 0, 0};
    __builtin_amdgcn_tensor_load_to_lds(g0, g1, z4, z4, z8, 0);
#else
    __builtin_amdgcn_tensor_load_to_lds(g0, g1, z4, z4, 0);
#endif
}
#endif // HAVE_TDM

// ---------------------------------------------------------------------------
// Shared WMMA GEMM core: c[4] += A_tile(16xK) * W_tile(64xK)^T, K = EE.
// Two-phase ping-pong: buffers written directly by loads, consumed one phase
// later -> no register rotation copies, loads a full phase ahead of WMMAs.
// ---------------------------------------------------------------------------
static __device__ inline void gemm_acc(const __bf16* __restrict__ arow,
                                       const __bf16* __restrict__ W,
                                       int jt, int ln, int hl, v8f c[4]) {
    const int koff = hl * 8;
#define LD_A(ek) (*(const v16bf*)(arow + (ek) + hl * 16))
#define LD_B(jj, ek) (*(const v16bf*)(W + (size_t)(jt + (jj) * 16 + ln) * EE + \
                                      (ek) + hl * 16))
    v16bf a0 = LD_A(0);
    v16bf b0[4];
#pragma unroll
    for (int jj = 0; jj < 4; ++jj) b0[jj] = LD_B(jj, 0);
    v16bf a1;
    v16bf b1[4];
    for (int ek = 0; ek < EE - 64; ek += 64) {
        a1 = LD_A(ek + 32);
#pragma unroll
        for (int jj = 0; jj < 4; ++jj) b1[jj] = LD_B(jj, ek + 32);
#pragma unroll
        for (int jj = 0; jj < 4; ++jj) c[jj] = wmma_bf16(a0, b0[jj], c[jj]);
        a0 = LD_A(ek + 64);
#pragma unroll
        for (int jj = 0; jj < 4; ++jj) b0[jj] = LD_B(jj, ek + 64);
#pragma unroll
        for (int jj = 0; jj < 4; ++jj) c[jj] = wmma_bf16(a1, b1[jj], c[jj]);
    }
    a1 = LD_A(EE - 32);
#pragma unroll
    for (int jj = 0; jj < 4; ++jj) b1[jj] = LD_B(jj, EE - 32);
#pragma unroll
    for (int jj = 0; jj < 4; ++jj) c[jj] = wmma_bf16(a0, b0[jj], c[jj]);
#pragma unroll
    for (int jj = 0; jj < 4; ++jj) c[jj] = wmma_bf16(a1, b1[jj], c[jj]);
#undef LD_A
#undef LD_B
}

// ---------------------------------------------------------------------------
// K0: LayerNorm (per token) + fp32->bf16
// ---------------------------------------------------------------------------
__global__ __launch_bounds__(256) void ln_kernel(const float* __restrict__ x,
                                                 const float* __restrict__ gamma,
                                                 const float* __restrict__ beta,
                                                 __bf16* __restrict__ xnbf) {
    const int i = blockIdx.x;                 // token
    const int t = threadIdx.x;                // 0..255, 4 elements each
    const float* row = x + (size_t)i * EE;
    float4 v = ((const float4*)row)[t];
    float s  = v.x + v.y + v.z + v.w;
    float s2 = v.x * v.x + v.y * v.y + v.z * v.z + v.w * v.w;
#pragma unroll
    for (int off = 16; off >= 1; off >>= 1) {
        s  += __shfl_xor(s,  off, 32);
        s2 += __shfl_xor(s2, off, 32);
    }
    __shared__ float ss[8], sq[8], stat[2];
    int wv = t >> 5;
    if ((t & 31) == 0) { ss[wv] = s; sq[wv] = s2; }
    __syncthreads();
    if (t == 0) {
        float a = 0.f, b2 = 0.f;
#pragma unroll
        for (int k = 0; k < 8; ++k) { a += ss[k]; b2 += sq[k]; }
        float mu  = a * (1.0f / EE);
        float var = b2 * (1.0f / EE) - mu * mu;
        stat[0] = mu;
        stat[1] = rsqrtf(var + 1e-5f);
    }
    __syncthreads();
    float mu = stat[0], rstd = stat[1];
    float4 g  = ((const float4*)gamma)[t];
    float4 be = ((const float4*)beta)[t];
    int e = t * 4;
    __bf16* drow = xnbf + (size_t)i * EE + e;
    drow[0] = f2bf((v.x - mu) * rstd * g.x + be.x);
    drow[1] = f2bf((v.y - mu) * rstd * g.y + be.y);
    drow[2] = f2bf((v.z - mu) * rstd * g.z + be.z);
    drow[3] = f2bf((v.w - mu) * rstd * g.w + be.w);
}

// ---------------------------------------------------------------------------
// K1: fp32 -> bf16 (weights), 4 elements/thread
// ---------------------------------------------------------------------------
__global__ __launch_bounds__(256) void cvt_kernel(const float* __restrict__ src,
                                                  __bf16* __restrict__ dst) {
    int idx4 = blockIdx.x * 256 + threadIdx.x;
    float4 v = ((const float4*)src)[idx4];
    __bf16* d = dst + (size_t)idx4 * 4;
    d[0] = f2bf(v.x); d[1] = f2bf(v.y); d[2] = f2bf(v.z); d[3] = f2bf(v.w);
}

// K1b: int mask -> float
__global__ __launch_bounds__(256) void maskcvt_kernel(const int* __restrict__ m,
                                                      float* __restrict__ mf) {
    int i = blockIdx.x * 256 + threadIdx.x;
    mf[i] = (float)m[i];
}

// ---------------------------------------------------------------------------
// K2: QKV projection GEMM (pipelined WMMA core).
// mode==0: row layout [B,H,L,D] (scale applied), mode==1: transposed [B,H,D,L]
// ---------------------------------------------------------------------------
__global__ __launch_bounds__(256) void qkv_gemm(const __bf16* __restrict__ A,
                                                const __bf16* __restrict__ W,
                                                const float* __restrict__ bvec,
                                                __bf16* __restrict__ dst,
                                                int mode, float scale) {
    const int lane = threadIdx.x & 31;
    const int wave = threadIdx.x >> 5;
    const int tile = blockIdx.x * 8 + wave;        // 4096 wave-tiles
    const int it = tile >> 4;                      // M tiles of 16
    const int jt = (tile & 15) << 6;               // N tiles of 64
    const int ln = lane & 15;
    const int hl = lane >> 4;
    const __bf16* arow = A + (size_t)(it * 16 + ln) * EE;

    v8f c[4];
#pragma unroll
    for (int jj = 0; jj < 4; ++jj) c[jj] = zero8();
    gemm_acc(arow, W, jt, ln, hl, c);

#pragma unroll
    for (int jj = 0; jj < 4; ++jj) {
#pragma unroll
        for (int g = 0; g < 8; ++g) {
            int i = it * 16 + g + hl * 8;
            int j = jt + jj * 16 + ln;
            float val = c[jj][g] + bvec[j];
            int bidx = i >> 10, tok = i & (LL - 1);
            int h = j >> 6, d = j & (DD - 1);
            if (mode)   // V transposed: [B,H,D,L]
                dst[(((size_t)bidx * HH + h) * DD + d) * LL + tok] = f2bf(val);
            else        // Q/K: [B,H,L,D]
                dst[(((size_t)bidx * HH + h) * LL + tok) * DD + d] = f2bf(val * scale);
        }
    }
}

// ---------------------------------------------------------------------------
// K3: flash attention, one wave per (b, h, 16-query tile).
// K/V/bias tiles staged into LDS via the Tensor Data Mover, double-buffered,
// synchronized with s_wait_tensorcnt. S^T = K_tile*Q^T (softmax rows in
// lanes); P^T built from S^T C-fragments with shfl_xor(16); out^T = V^T*P^T.
// ---------------------------------------------------------------------------
__global__ __launch_bounds__(32) void attn_kernel(const __bf16* __restrict__ Qb,
                                                  const __bf16* __restrict__ Kb,
                                                  const __bf16* __restrict__ Vt,
                                                  const float* __restrict__ bias,
                                                  const float* __restrict__ maskf,
                                                  const float* __restrict__ gate,
                                                  __bf16* __restrict__ Obuf) {
    const int lane = threadIdx.x;
    const int qt = blockIdx.x, h = blockIdx.y, b = blockIdx.z;
    const int ln = lane & 15;
    const int hl = lane >> 4;
    const int koff = hl * 8;
    const size_t bh = (size_t)b * HH + h;
    const __bf16* Qp = Qb + bh * LL * DD;
    const __bf16* Kp = Kb + bh * LL * DD;
    const __bf16* Vp = Vt + bh * DD * LL;
    const int q = qt * 16 + ln;
    const float* bq0 = bias + bh * (size_t)LL * LL + (size_t)(qt * 16) * LL; // tile base
    const float* mkp = maskf + b * LL;
    const float gh = gate[h];

    // Q^T B-fragments (lane = query column, d = 8h+s and 16+8h+s per lane-half)
    v16bf qb0 = mk16(*(const v8bf*)(Qp + (size_t)q * DD + koff), *(const v8bf*)(Qp + (size_t)q * DD + 16 + koff));
    v16bf qb1 = mk16(*(const v8bf*)(Qp + (size_t)q * DD + 32 + koff), *(const v8bf*)(Qp + (size_t)q * DD + 48 + koff));

    __shared__ __bf16 smK[2][32 * DD];   // 32 keys x 64 d   (4 KB each)
    __shared__ __bf16 smV[2][DD * 32];   // 64 d   x 32 keys (4 KB each)
    __shared__ float  smB[2][16 * 32];   // 16 q   x 32 keys (2 KB each)

#if HAVE_TDM
    // TDM: 3 descriptors per 32-key tile (K, V^T, bias), dims in dwords
    #define ISSUE_TILE(kb_, buf_)                                                     \
        do {                                                                          \
            tdm_load(tdm_g0(ldsaddr(&smK[buf_][0]),                                   \
                            (unsigned long long)(uintptr_t)(Kp + (size_t)(kb_) * DD)),\
                     tdm_g1(32, 32, 32, 32, 32));                                     \
            tdm_load(tdm_g0(ldsaddr(&smV[buf_][0]),                                   \
                            (unsigned long long)(uintptr_t)(Vp + (kb_))),             \
                     tdm_g1(16, 64, 16, 64, LL / 2));                                 \
            tdm_load(tdm_g0(ldsaddr(&smB[buf_][0]),                                   \
                            (unsigned long long)(uintptr_t)(bq0 + (kb_))),            \
                     tdm_g1(32, 16, 32, 16, LL));                                     \
        } while (0)
#else
    // fallback: plain per-lane global->LDS copies (32 lanes)
    #define ISSUE_TILE(kb_, buf_)                                                     \
        do {                                                                          \
            for (int r = lane; r < 32; r += 32) {                                     \
                ((ulonglong2*)&smK[buf_][r * DD])[0] =                                \
                    ((const ulonglong2*)(Kp + (size_t)((kb_) + r) * DD))[0];          \
                ((ulonglong2*)&smK[buf_][r * DD])[1] =                                \
                    ((const ulonglong2*)(Kp + (size_t)((kb_) + r) * DD))[1];          \
                ((ulonglong2*)&smK[buf_][r * DD])[2] =                                \
                    ((const ulonglong2*)(Kp + (size_t)((kb_) + r) * DD))[2];          \
                ((ulonglong2*)&smK[buf_][r * DD])[3] =                                \
                    ((const ulonglong2*)(Kp + (size_t)((kb_) + r) * DD))[3];          \
            }                                                                         \
            for (int r = lane; r < DD; r += 32) {                                     \
                ((ulonglong2*)&smV[buf_][r * 32])[0] =                                \
                    ((const ulonglong2*)(Vp + (size_t)r * LL + (kb_)))[0];            \
                ((ulonglong2*)&smV[buf_][r * 32])[1] =                                \
                    ((const ulonglong2*)(Vp + (size_t)r * LL + (kb_)))[1];            \
            }                                                                         \
            for (int r = lane; r < 16; r += 32) {                                     \
                ((float4*)&smB[buf_][r * 32])[0] =                                    \
                    ((const float4*)(bq0 + (size_t)r * LL + (kb_)))[0];               \
                ((float4*)&smB[buf_][r * 32])[1] =                                    \
                    ((const float4*)(bq0 + (size_t)r * LL + (kb_)))[1];               \
                ((float4*)&smB[buf_][r * 32])[2] =                                    \
                    ((const float4*)(bq0 + (size_t)r * LL + (kb_)))[2];               \
                ((float4*)&smB[buf_][r * 32])[3] =                                    \
                    ((const float4*)(bq0 + (size_t)r * LL + (kb_)))[3];               \
            }                                                                         \
        } while (0)
#endif

    v8f o[4];
#pragma unroll
    for (int c = 0; c < 4; ++c) o[c] = zero8();
    float mrun = -1e30f, lsum = 0.0f;

    ISSUE_TILE(0, 0);
    int buf = 0;

    for (int kb = 0; kb < LL; kb += 32) {
        if (kb + 32 < LL) {
            ISSUE_TILE(kb + 32, buf ^ 1);
#if HAVE_TDM
            __builtin_amdgcn_s_wait_tensorcnt((short)3); // current tile's 3 done
#endif
        } else {
#if HAVE_TDM
            __builtin_amdgcn_s_wait_tensorcnt((short)0);
#endif
        }
        __builtin_prefetch(mkp + kb + 32, 0, 0);

        // S^T fragments: s0 -> keys kb..kb+15, s1 -> keys kb+16..kb+31
        v8f s0 = zero8(), s1 = zero8();
        {
            const __bf16* kr = &smK[buf][ln * DD];
            v16bf a0 = mk16(*(const v8bf*)(kr + koff),      *(const v8bf*)(kr + 16 + koff));
            v16bf a1 = mk16(*(const v8bf*)(kr + 32 + koff), *(const v8bf*)(kr + 48 + koff));
            s0 = wmma_bf16(a0, qb0, s0);
            s0 = wmma_bf16(a1, qb1, s0);
        }
        {
            const __bf16* kr = &smK[buf][(16 + ln) * DD];
            v16bf a0 = mk16(*(const v8bf*)(kr + koff),      *(const v8bf*)(kr + 16 + koff));
            v16bf a1 = mk16(*(const v8bf*)(kr + 32 + koff), *(const v8bf*)(kr + 48 + koff));
            s1 = wmma_bf16(a0, qb0, s1);
            s1 = wmma_bf16(a1, qb1, s1);
        }
        // gate*bias (from LDS) + key mask
        const float* b0p = &smB[buf][ln * 32 + hl * 8];
        const float* b1p = b0p + 16;
        const float* m0p = mkp + kb + hl * 8;
        const float* m1p = mkp + kb + 16 + hl * 8;
        float p0[8], p1[8];
        float tmax = -1e30f;
#pragma unroll
        for (int g = 0; g < 8; ++g) {
            float v0 = s0[g] + gh * b0p[g] + (1.0f - m0p[g]) * (-10000.0f);
            float v1 = s1[g] + gh * b1p[g] + (1.0f - m1p[g]) * (-10000.0f);
            p0[g] = v0; p1[g] = v1;
            tmax = fmaxf(tmax, fmaxf(v0, v1));
        }
        tmax = fmaxf(tmax, __shfl_xor(tmax, 16, 32));
        float newm  = fmaxf(mrun, tmax);
        float alpha = exp2f((mrun - newm) * LOG2E);
        float psum = 0.0f;
#pragma unroll
        for (int g = 0; g < 8; ++g) {
            p0[g] = exp2f((p0[g] - newm) * LOG2E);
            p1[g] = exp2f((p1[g] - newm) * LOG2E);
            psum += p0[g] + p1[g];
        }
        psum += __shfl_xor(psum, 16, 32);
        lsum = lsum * alpha + psum;
        mrun = newm;
#pragma unroll
        for (int c = 0; c < 4; ++c)
#pragma unroll
            for (int g = 0; g < 8; ++g) o[c][g] *= alpha;

        // Build P^T B-fragment: lane-half exchange via shfl_xor(16)
        v16bf pb;
#pragma unroll
        for (int i2 = 0; i2 < 8; ++i2) {
            float e0x = __shfl_xor(p0[i2], 16, 32);
            float e1x = __shfl_xor(p1[i2], 16, 32);
            pb[i2]     = f2bf(hl == 0 ? p0[i2] : e1x);
            pb[i2 + 8] = f2bf(hl == 0 ? e0x    : p1[i2]);
        }
        // out^T += V^T * P^T  (V tile in LDS: [64 d][32 keys])
#pragma unroll
        for (int c = 0; c < 4; ++c) {
            const __bf16* vr = &smV[buf][(c * 16 + ln) * 32];
            v16bf av = *(const v16bf*)(vr + hl * 16);
            o[c] = wmma_bf16(av, pb, o[c]);
        }
        buf ^= 1;
    }
    // epilogue: 1/lsum, query mask, write [B,L,E] bf16 for the Wo GEMM
    float qm  = maskf[b * LL + q];
    float inv = (lsum > 0.0f) ? (1.0f / lsum) : 0.0f;
    float f = qm * inv;
    __bf16* orow = Obuf + ((size_t)(b * LL + q)) * EE + h * DD;
#pragma unroll
    for (int c = 0; c < 4; ++c)
#pragma unroll
        for (int g = 0; g < 8; ++g) {
            int d = c * 16 + g + hl * 8;
            orow[d] = f2bf(o[c][g] * f);
        }
    #undef ISSUE_TILE
}

// ---------------------------------------------------------------------------
// K4: output projection + residual:  out = x + Obuf @ Wo^T + bo   (fp32 out)
// ---------------------------------------------------------------------------
__global__ __launch_bounds__(256) void out_gemm(const __bf16* __restrict__ A,
                                                const __bf16* __restrict__ W,
                                                const float* __restrict__ bo,
                                                const float* __restrict__ xres,
                                                float* __restrict__ out) {
    const int lane = threadIdx.x & 31;
    const int wave = threadIdx.x >> 5;
    const int tile = blockIdx.x * 8 + wave;
    const int it = tile >> 4;
    const int jt = (tile & 15) << 6;
    const int ln = lane & 15;
    const int hl = lane >> 4;
    const __bf16* arow = A + (size_t)(it * 16 + ln) * EE;

    v8f c[4];
#pragma unroll
    for (int jj = 0; jj < 4; ++jj) c[jj] = zero8();
    gemm_acc(arow, W, jt, ln, hl, c);

#pragma unroll
    for (int jj = 0; jj < 4; ++jj) {
#pragma unroll
        for (int g = 0; g < 8; ++g) {
            int i = it * 16 + g + hl * 8;
            int j = jt + jj * 16 + ln;
            out[(size_t)i * EE + j] = xres[(size_t)i * EE + j] + bo[j] + c[jj][g];
        }
    }
}

// ---------------------------------------------------------------------------
extern "C" void kernel_launch(void* const* d_in, const int* in_sizes, int n_in,
                              void* d_out, int out_size, void* d_ws, size_t ws_size,
                              hipStream_t stream) {
    const float* x     = (const float*)d_in[0];
    const float* bias  = (const float*)d_in[1];
    const int*   mask  = (const int*)  d_in[2];
    const float* Wq    = (const float*)d_in[3];
    const float* bq    = (const float*)d_in[4];
    const float* Wk    = (const float*)d_in[5];
    const float* bk    = (const float*)d_in[6];
    const float* Wv    = (const float*)d_in[7];
    const float* bv    = (const float*)d_in[8];
    const float* Wo    = (const float*)d_in[9];
    const float* bo    = (const float*)d_in[10];
    const float* gamma = (const float*)d_in[11];
    const float* beta  = (const float*)d_in[12];
    const float* gate  = (const float*)d_in[13];
    float* out = (float*)d_out;

    char* ws = (char*)d_ws;
    const size_t MB = 1024 * 1024;
    __bf16* xnbf = (__bf16*)(ws + 0);          // 8 MB
    __bf16* wqbf = (__bf16*)(ws + 8  * MB);    // 2 MB each
    __bf16* wkbf = (__bf16*)(ws + 10 * MB);
    __bf16* wvbf = (__bf16*)(ws + 12 * MB);
    __bf16* wobf = (__bf16*)(ws + 14 * MB);
    __bf16* Qbf  = (__bf16*)(ws + 16 * MB);    // 8 MB
    __bf16* Kbf  = (__bf16*)(ws + 24 * MB);    // 8 MB
    __bf16* Vtb  = (__bf16*)(ws + 32 * MB);    // 8 MB
    __bf16* Obuf = (__bf16*)(ws + 40 * MB);    // 8 MB
    float*  mkf  = (float*) (ws + 48 * MB);    // 16 KB

    // LayerNorm + bf16
    ln_kernel<<<MM, 256, 0, stream>>>(x, gamma, beta, xnbf);
    // weight / mask conversions
    cvt_kernel<<<(EE * EE) / 1024, 256, 0, stream>>>(Wq, wqbf);
    cvt_kernel<<<(EE * EE) / 1024, 256, 0, stream>>>(Wk, wkbf);
    cvt_kernel<<<(EE * EE) / 1024, 256, 0, stream>>>(Wv, wvbf);
    cvt_kernel<<<(EE * EE) / 1024, 256, 0, stream>>>(Wo, wobf);
    maskcvt_kernel<<<MM / 256, 256, 0, stream>>>(mask, mkf);
    // projections (scale folded into Q)
    qkv_gemm<<<512, 256, 0, stream>>>(xnbf, wqbf, bq, Qbf, 0, 0.125f);
    qkv_gemm<<<512, 256, 0, stream>>>(xnbf, wkbf, bk, Kbf, 0, 1.0f);
    qkv_gemm<<<512, 256, 0, stream>>>(xnbf, wvbf, bv, Vtb, 1, 1.0f);
    // flash attention (TDM-staged tiles)
    dim3 agrid(LL / 16, HH, BB);
    attn_kernel<<<agrid, 32, 0, stream>>>(Qbf, Kbf, Vtb, bias, mkf, gate, Obuf);
    // output projection + residual
    out_gemm<<<512, 256, 0, stream>>>(Obuf, wobf, bo, x, out);
}
